// LongHorizonForwardNet_64501818851843
// MI455X (gfx1250) — matrix operand order repaired and checked by reading, not compiled, not run on hardware
//
#include <hip/hip_runtime.h>

typedef __attribute__((ext_vector_type(16))) _Float16 v16h;
typedef __attribute__((ext_vector_type(8)))  _Float16 v8h;
typedef __attribute__((ext_vector_type(8)))  float    v8f;

#define S_STEPS 8
#define BATCH   32768
#define FEAT    512
#define ASIZE   12
#define HIDDEN  140
#define INSZ    (FEAT + ASIZE)      /* 524 */
#define K1PAD   544                 /* 17 * 32 */
#define N1PAD   144                 /* 9  * 16 */
#define K2PAD   160                 /* 5  * 32 */
#define XSTR    552                 /* LDS f16 row stride for X (bank-friendly) */
#define HSTR    168                 /* LDS f16 row stride for H (bank-friendly) */
#define MTILE   128
#define BSTR1   136                 /* staged W1 panel row stride: 4kt*32 + 8 pad */
#define BSTR2   168                 /* staged W2 panel row stride: 160 + 8 pad */
#define BUFE    21504               /* f16 elems per staging buffer = 128*168 */
#define W1H_ELEMS (S_STEPS * N1PAD * K1PAD)   /* 626688 */
#define W2H_ELEMS (S_STEPS * FEAT  * K2PAD)   /* 655360 */

union F16x16 { v16h v; v8h h[2]; };

// ---------------------------------------------------------------------------
// Prep: f32 weights -> f16, transposed [n][k], zero-padded, so one WMMA
// B-fragment is two runs of 8 contiguous f16 per lane.
// ---------------------------------------------------------------------------
__global__ void prep_weights(const float* __restrict__ W1,
                             const float* __restrict__ W2,
                             _Float16* __restrict__ w1h,
                             _Float16* __restrict__ w2h) {
  int i = blockIdx.x * blockDim.x + threadIdx.x;
  if (i < W1H_ELEMS) {
    int s = i / (N1PAD * K1PAD);
    int r = i % (N1PAD * K1PAD);
    int n = r / K1PAD, k = r % K1PAD;
    float v = (n < HIDDEN && k < INSZ) ? W1[((size_t)s * INSZ + k) * HIDDEN + n] : 0.f;
    w1h[i] = (_Float16)v;
  } else if (i < W1H_ELEMS + W2H_ELEMS) {
    int j = i - W1H_ELEMS;
    int s = j / (FEAT * K2PAD);
    int r = j % (FEAT * K2PAD);
    int n = r / K2PAD, k = r % K2PAD;
    float v = (k < HIDDEN) ? W2[((size_t)s * HIDDEN + k) * FEAT + n] : 0.f;
    w2h[j] = (_Float16)v;
  }
}

// ---------------------------------------------------------------------------
// Main: one block = 128 batch rows, resident in LDS (f16) across all 8 steps.
// 8 waves x 16 rows. Weight panels double-buffered through LDS (8x L2 traffic
// reduction). All matrix math: v_wmma_f32_16x16x32_f16 (f32 accumulate).
// ---------------------------------------------------------------------------
__global__ __launch_bounds__(256, 1)
void fwdnet_kernel(const float* __restrict__ feat_in,
                   const float* __restrict__ b1,
                   const float* __restrict__ b2,
                   const int*   __restrict__ actions,
                   const _Float16* __restrict__ w1h,
                   const _Float16* __restrict__ w2h,
                   float* __restrict__ out) {
  extern __shared__ _Float16 sm[];
  _Float16* X  = sm;                              // [128][XSTR] feat + one-hot
  _Float16* H  = X + MTILE * XSTR;                // [128][HSTR] hidden acts
  _Float16* B0 = H + MTILE * HSTR;                // staging buffer 0
  _Float16* B1 = B0 + BUFE;                       // staging buffer 1

  const int tid  = threadIdx.x;
  const int wave = tid >> 5;                      // wave32
  const int lane = tid & 31;
  const int lo   = lane & 15;
  const int hi   = lane >> 4;
  const int r0   = blockIdx.x * MTILE;
  const int xrow = wave * 16 + lo;

  uint4 st[10];                                   // staging registers

  // ---- init: X[:,0:512] <- features (f32->f16); H <- 0 --------------------
  for (int i = tid; i < MTILE * (FEAT / 4); i += 256) {
    int row = i >> 7, c4 = i & 127;
    float4 f = ((const float4*)(feat_in + (size_t)(r0 + row) * FEAT))[c4];
    _Float16* xp = X + row * XSTR + c4 * 4;
    xp[0] = (_Float16)f.x; xp[1] = (_Float16)f.y;
    xp[2] = (_Float16)f.z; xp[3] = (_Float16)f.w;
  }
  for (int i = tid; i < (MTILE * HSTR) / 2; i += 256)
    ((unsigned int*)H)[i] = 0u;

  for (int s = 0; s < S_STEPS; ++s) {
    const _Float16* w1s = w1h + (size_t)s * N1PAD * K1PAD;
    const _Float16* w2s = w2h + (size_t)s * FEAT * K2PAD;

    // ---- one-hot columns X[:,512:544] -----------------------------------
    if (tid < MTILE) {
      int n = r0 + tid;
      _Float16* xr = X + tid * XSTR + FEAT;
#pragma unroll
      for (int c = 0; c < 32; ++c) xr[c] = (_Float16)0.f;
      int a0 = actions[((size_t)s * 2 + 0) * BATCH + n];
      int a1 = actions[((size_t)s * 2 + 1) * BATCH + n];
      xr[a0]     = (_Float16)1.f;
      xr[5 + a1] = (_Float16)1.f;
    }

    // ---- GEMM1: H = leaky_relu(X @ W1 + b1), staged/pipelined ------------
    v8f acc1[9];
#pragma unroll
    for (int t = 0; t < 9; ++t) acc1[t] = (v8f){0,0,0,0,0,0,0,0};

    // prefetch chunk 0 (kt 0..3) into B0
#pragma unroll
    for (int i = 0; i < 9; ++i) {
      int u = tid + i * 256;
      if (u < N1PAD * 16) {
        int n = u >> 4, k8 = u & 15;
        st[i] = *(const uint4*)(w1s + n * K1PAD + k8 * 8);
      }
    }
#pragma unroll
    for (int i = 0; i < 9; ++i) {
      int u = tid + i * 256;
      if (u < N1PAD * 16) {
        int n = u >> 4, k8 = u & 15;
        *(uint4*)(B0 + n * BSTR1 + k8 * 8) = st[i];
      }
    }
    __syncthreads();   // one-hot + chunk0 visible

#pragma unroll 1
    for (int c = 0; c < 5; ++c) {               // k-chunks: 4,4,4,4,1 tiles
      const int kt0   = c * 4;
      const int kcnt  = (c == 4) ? 1 : 4;
      _Float16* bcur  = (c & 1) ? B1 : B0;
      _Float16* bnxt  = (c & 1) ? B0 : B1;
      const int kcntn = (c + 1 == 4) ? 1 : 4;

      if (c + 1 < 5) {                          // issue next-chunk loads
#pragma unroll
        for (int i = 0; i < 9; ++i) {
          int u = tid + i * 256;
          if (u < N1PAD * kcntn * 4) {
            int n = u / (kcntn * 4), k8 = u % (kcntn * 4);
            st[i] = *(const uint4*)(w1s + n * K1PAD + (kt0 + 4) * 32 + k8 * 8);
          }
        }
      }
#pragma unroll 1
      for (int ktl = 0; ktl < kcnt; ++ktl) {    // compute current chunk
        F16x16 a;
        const _Float16* ap = X + xrow * XSTR + (kt0 + ktl) * 32 + hi * 8;
        a.h[0] = *(const v8h*)ap;
        a.h[1] = *(const v8h*)(ap + 16);
#pragma unroll
        for (int nt = 0; nt < 9; ++nt) {
          F16x16 b;
          const _Float16* bp = bcur + (nt * 16 + lo) * BSTR1 + ktl * 32 + hi * 8;
          b.h[0] = *(const v8h*)bp;
          b.h[1] = *(const v8h*)(bp + 16);
          acc1[nt] = __builtin_amdgcn_wmma_f32_16x16x32_f16(
              false, a.v, false, b.v, (short)0, acc1[nt], false, false);
        }
      }
      if (c + 1 < 5) {                          // commit next chunk to LDS
#pragma unroll
        for (int i = 0; i < 9; ++i) {
          int u = tid + i * 256;
          if (u < N1PAD * kcntn * 4) {
            int n = u / (kcntn * 4), k8 = u % (kcntn * 4);
            *(uint4*)(bnxt + n * BSTR1 + k8 * 8) = st[i];
          }
        }
      }
      __syncthreads();
    }
    // epilogue: bias + leaky relu -> H (f16); intra-wave rows only
#pragma unroll
    for (int nt = 0; nt < 9; ++nt) {
      int col = nt * 16 + lo;
      float bias = (col < HIDDEN) ? b1[s * HIDDEN + col] : 0.f;
#pragma unroll
      for (int v = 0; v < 8; ++v) {
        float x = acc1[nt][v] + bias;
        x = x > 0.f ? x : 0.1f * x;
        H[(wave * 16 + v + 8 * hi) * HSTR + col] = (_Float16)x;
      }
    }

    // ---- GEMM2: OUT = H @ W2 + b2 -> d_out, X (next feat); staged --------
    F16x16 a2[5];
#pragma unroll
    for (int kt = 0; kt < 5; ++kt) {
      const _Float16* ap = H + xrow * HSTR + kt * 32 + hi * 8;
      a2[kt].h[0] = *(const v8h*)ap;
      a2[kt].h[1] = *(const v8h*)(ap + 16);
    }
    float* outs = out + (size_t)s * BATCH * FEAT;

    // prefetch W2 chunk 0 (cols 0..127, full K=160) into B0
#pragma unroll
    for (int i = 0; i < 10; ++i) {              // 10*256 == 128*20 exactly
      int u = tid + i * 256;
      int n = u / 20, k8 = u % 20;
      st[i] = *(const uint4*)(w2s + n * K2PAD + k8 * 8);
    }
#pragma unroll
    for (int i = 0; i < 10; ++i) {
      int u = tid + i * 256;
      int n = u / 20, k8 = u % 20;
      *(uint4*)(B0 + n * BSTR2 + k8 * 8) = st[i];
    }
    __syncthreads();

#pragma unroll 1
    for (int c = 0; c < 4; ++c) {               // 4 chunks of 128 out-cols
      _Float16* bcur = (c & 1) ? B1 : B0;
      _Float16* bnxt = (c & 1) ? B0 : B1;

      if (c + 1 < 4) {                          // issue next-chunk loads
#pragma unroll
        for (int i = 0; i < 10; ++i) {
          int u = tid + i * 256;
          int n = u / 20, k8 = u % 20;
          st[i] = *(const uint4*)(w2s + ((c + 1) * 128 + n) * K2PAD + k8 * 8);
        }
      }
      v8f acc2[8];
#pragma unroll
      for (int t = 0; t < 8; ++t) acc2[t] = (v8f){0,0,0,0,0,0,0,0};
#pragma unroll 1
      for (int kt = 0; kt < 5; ++kt) {
#pragma unroll
        for (int nt = 0; nt < 8; ++nt) {
          F16x16 b;
          const _Float16* bp = bcur + (nt * 16 + lo) * BSTR2 + kt * 32 + hi * 8;
          b.h[0] = *(const v8h*)bp;
          b.h[1] = *(const v8h*)(bp + 16);
          acc2[nt] = __builtin_amdgcn_wmma_f32_16x16x32_f16(
              false, a2[kt].v, false, b.v, (short)0, acc2[nt], false, false);
        }
      }
      // epilogue: bias, f32 prediction store, f16 carry into X
#pragma unroll
      for (int nt = 0; nt < 8; ++nt) {
        int col = c * 128 + nt * 16 + lo;
        float bias = b2[s * FEAT + col];
#pragma unroll
        for (int v = 0; v < 8; ++v) {
          int m = wave * 16 + v + 8 * hi;
          float y = acc2[nt][v] + bias;
          outs[(size_t)(r0 + m) * FEAT + col] = y;
          X[m * XSTR + col] = (_Float16)y;
        }
      }
      if (c + 1 < 4) {                          // commit next chunk to LDS
#pragma unroll
        for (int i = 0; i < 10; ++i) {
          int u = tid + i * 256;
          int n = u / 20, k8 = u % 20;
          *(uint4*)(bnxt + n * BSTR2 + k8 * 8) = st[i];
        }
      }
      __syncthreads();
    }
    // final barrier of chunk loop orders X carry vs next step's reads
  }
}

extern "C" void kernel_launch(void* const* d_in, const int* in_sizes, int n_in,
                              void* d_out, int out_size, void* d_ws, size_t ws_size,
                              hipStream_t stream) {
  const float* feat    = (const float*)d_in[0];
  const float* W1      = (const float*)d_in[1];
  const float* b1      = (const float*)d_in[2];
  const float* W2      = (const float*)d_in[3];
  const float* b2      = (const float*)d_in[4];
  const int*   actions = (const int*)d_in[5];

  _Float16* w1h = (_Float16*)d_ws;
  _Float16* w2h = w1h + W1H_ELEMS;

  int total = W1H_ELEMS + W2H_ELEMS;
  prep_weights<<<(total + 255) / 256, 256, 0, stream>>>(W1, W2, w1h, w2h);

  size_t smem = (size_t)(MTILE * XSTR + MTILE * HSTR + 2 * BUFE) * sizeof(_Float16); // 270336 B
  (void)hipFuncSetAttribute((const void*)fwdnet_kernel,
                            hipFuncAttributeMaxDynamicSharedMemorySize,
                            (int)smem);
  fwdnet_kernel<<<BATCH / MTILE, 256, smem, stream>>>(
      feat, b1, b2, actions, w1h, w2h, (float*)d_out);
}
